// ResidualVectorQuantizer_47347719471383
// MI455X (gfx1250) — matrix operand order repaired and checked by reading, not compiled, not run on hardware
//
#include <hip/hip_runtime.h>

typedef __attribute__((ext_vector_type(16))) __bf16 v16bf;
typedef __attribute__((ext_vector_type(8)))  float  v8f;

#define RQ 8
#define RK 1024
#define RD 128
#define RB 16
#define RT 2048
#define RN (RB * RT)           // 32768 tokens
#define RBETA 0.25f

// workspace layout (bytes)
#define WS_RES   0                               // residual fp32 [N, D]   = 16 MB
#define WS_CBHI  (RN * RD * 4)                   // bf16 hi [Q*K*D]        =  2 MB
#define WS_CBLO  (WS_CBHI + RQ * RK * RD * 2)    // bf16 lo [Q*K*D]        =  2 MB
#define WS_C2    (WS_CBLO + RQ * RK * RD * 2)    // c2 fp32 [Q*K]          = 32 KB
#define WS_PART  (WS_C2 + RQ * RK * 4)           // loss partials [Q*256]  =  8 KB

__device__ __forceinline__ void wait_async_zero() {
#if defined(__has_builtin)
#if __has_builtin(__builtin_amdgcn_s_wait_asynccnt)
  __builtin_amdgcn_s_wait_asynccnt(0);
  return;
#else
  asm volatile("s_wait_asynccnt 0x0" ::: "memory");
  return;
#endif
#else
  asm volatile("s_wait_asynccnt 0x0" ::: "memory");
#endif
}

// ---------------------------------------------------------------------------
// Transpose x[B, D, T] -> residual[N=B*T, D] (x reads coalesced).
__global__ __launch_bounds__(256)
void rvq_prep_res(const float* __restrict__ x, float* __restrict__ res) {
  unsigned gid = blockIdx.x * 256u + threadIdx.x;        // < RN*RD
  unsigned b   = gid >> 18;                              // / (D*T = 262144)
  unsigned rem = gid & 262143u;
  unsigned d   = rem >> 11;                              // / T
  unsigned t   = rem & 2047u;
  unsigned n   = b * RT + t;
  res[(size_t)n * RD + d] = x[gid];
}

// ---------------------------------------------------------------------------
// Split codebooks to bf16 hi/lo and compute exact fp32 c2[k] = sum_d c^2.
// One wave per codebook row (Q*K rows).
__global__ __launch_bounds__(256)
void rvq_prep_cb(const float* __restrict__ cb, __bf16* __restrict__ hi,
                 __bf16* __restrict__ lo, float* __restrict__ c2) {
  unsigned gid  = blockIdx.x * 256u + threadIdx.x;
  unsigned row  = gid >> 5;
  unsigned lane = gid & 31u;
  const float4 v = *(const float4*)(cb + (size_t)row * RD + lane * 4);
  float xs[4] = {v.x, v.y, v.z, v.w};
  float s = 0.f;
#pragma unroll
  for (int j = 0; j < 4; ++j) {
    float  xv = xs[j];
    __bf16 h  = (__bf16)xv;
    hi[(size_t)row * RD + lane * 4 + j] = h;
    lo[(size_t)row * RD + lane * 4 + j] = (__bf16)(xv - (float)h);
    s += xv * xv;
  }
#pragma unroll
  for (int off = 1; off < 32; off <<= 1) s += __shfl_xor(s, off, 32);
  if (lane == 0) c2[row] = s;
}

// ---------------------------------------------------------------------------
// One RVQ stage: WMMA distance GEMM + argmin + residual update + exact loss.
// Block = 256 threads = 8 waves; each wave owns a 16-token tile; grid = N/128.
// Codebook tiles are double-buffered in LDS via GLOBAL_LOAD_ASYNC_TO_LDS_B128.
__global__ __launch_bounds__(256)
void rvq_stage(float* __restrict__ residual,
               const __bf16* __restrict__ cb_hi_all,
               const __bf16* __restrict__ cb_lo_all,
               const float* __restrict__ c2_all,
               const float* __restrict__ cb_f32,   // original fp32 codebooks
               float* __restrict__ partials,       // [Q*256] per-block loss sums
               float* __restrict__ idx_out,        // d_out indices region, [B,Q,T] as float
               int stage) {
  // LDS tile: 16 codes x 128 dims, hi + lo, row stride 144 (pad kills conflicts),
  // double buffered for the async pipeline.
  __shared__ __attribute__((aligned(32))) __bf16 sh_hi[2][16 * 144];
  __shared__ __attribute__((aligned(32))) __bf16 sh_lo[2][16 * 144];
  __shared__ float wsum[8];

  const int tid  = threadIdx.x;
  const int wave = tid >> 5;
  const int lane = tid & 31;
  const int half = lane >> 4;         // 0: lanes 0-15, 1: lanes 16-31
  const int l15  = lane & 15;
  const int m0   = blockIdx.x * 128 + wave * 16;   // wave's first token

  const __bf16* cbh = cb_hi_all + (size_t)stage * RK * RD;
  const __bf16* cbl = cb_lo_all + (size_t)stage * RK * RD;
  const float*  c2s = c2_all + stage * RK;

  // ---- Load A fragments (16 tokens x 128 dims), contiguous k order (same as B):
  // lane l (M = l&15): elem j<8  -> K = 32f + 16*half + j
  //                    elem j>=8 -> K = 32f + 16*half + 8 + (j-8)
  v16bf a_hi[4], a_lo[4];
  {
    const float* rrow = residual + (size_t)(m0 + l15) * RD;
#pragma unroll
    for (int f = 0; f < 4; ++f) {
      int d0 = 32 * f + 16 * half;
      int d1 = d0 + 8;
#pragma unroll
      for (int j = 0; j < 8; ++j) {
        float  x0 = rrow[d0 + j];
        __bf16 h0 = (__bf16)x0;
        a_hi[f][j]     = h0;
        a_lo[f][j]     = (__bf16)(x0 - (float)h0);
        float  x1 = rrow[d1 + j];
        __bf16 h1 = (__bf16)x1;
        a_hi[f][j + 8] = h1;
        a_lo[f][j + 8] = (__bf16)(x1 - (float)h1);
      }
    }
  }

  float bv[8];
  int   bi[8];
#pragma unroll
  for (int v = 0; v < 8; ++v) { bv[v] = 3.0e38f; bi[v] = 0; }

  // cooperative async-loader coordinates: 256 threads x 16B per array per tile
  const int ldr = tid >> 4;           // code row 0..15 in tile
  const int ldc = (tid & 15) * 8;     // dim column
  // LDS byte offsets (generic LDS pointer low 32 bits == LDS byte address)
  unsigned ldsh[2] = {(unsigned)(uintptr_t)&sh_hi[0][ldr * 144 + ldc],
                      (unsigned)(uintptr_t)&sh_hi[1][ldr * 144 + ldc]};
  unsigned ldsl[2] = {(unsigned)(uintptr_t)&sh_lo[0][ldr * 144 + ldc],
                      (unsigned)(uintptr_t)&sh_lo[1][ldr * 144 + ldc]};

  auto issue_tile = [&](int kt, int buf) {
    unsigned long long gh =
        (unsigned long long)(uintptr_t)(cbh + (size_t)(kt * 16 + ldr) * RD + ldc);
    unsigned long long gl =
        (unsigned long long)(uintptr_t)(cbl + (size_t)(kt * 16 + ldr) * RD + ldc);
    asm volatile("global_load_async_to_lds_b128 %0, %1, off"
                 :: "v"(ldsh[buf]), "v"(gh) : "memory");
    asm volatile("global_load_async_to_lds_b128 %0, %1, off"
                 :: "v"(ldsl[buf]), "v"(gl) : "memory");
  };

  issue_tile(0, 0);   // prologue: tile 0 -> buffer 0

  for (int kt = 0; kt < 64; ++kt) {
    const int n0  = kt * 16;
    const int buf = kt & 1;
    if (kt + 2 < 64) {  // warm L2 two tiles ahead -> global_prefetch_b8
      __builtin_prefetch(cbh + (size_t)(n0 + 32 + ldr) * RD + ldc, 0, 1);
      __builtin_prefetch(cbl + (size_t)(n0 + 32 + ldr) * RD + ldc, 0, 1);
    }
    wait_async_zero();   // this wave's tile-kt async copies landed in LDS
    __syncthreads();     // all waves' copies landed; prior compute done
    if (kt + 1 < 64) issue_tile(kt + 1, buf ^ 1);  // overlap next fetch

    // B 32x16 layout: lane l (N = l&15): elem j -> K = 32f + 16*half + j (contiguous)
    v8f acc = {0.f, 0.f, 0.f, 0.f, 0.f, 0.f, 0.f, 0.f};
#pragma unroll
    for (int f = 0; f < 4; ++f) {
      const int off = l15 * 144 + 32 * f + 16 * half;   // *2B: 32B-aligned
      v16bf bh = *(const v16bf*)&sh_hi[buf][off];
      v16bf bl = *(const v16bf*)&sh_lo[buf][off];
      acc = __builtin_amdgcn_wmma_f32_16x16x32_bf16(false, a_hi[f], false, bh,
                                                    (short)0, acc, false, false);
      acc = __builtin_amdgcn_wmma_f32_16x16x32_bf16(false, a_lo[f], false, bh,
                                                    (short)0, acc, false, false);
      acc = __builtin_amdgcn_wmma_f32_16x16x32_bf16(false, a_hi[f], false, bl,
                                                    (short)0, acc, false, false);
    }

    // dist = ||c||^2 - 2 r.c  (||r||^2 constant per row -> irrelevant to argmin)
    float c2v = c2s[n0 + l15];
#pragma unroll
    for (int v = 0; v < 8; ++v) {       // C layout: lane holds M = v + 8*half, N = l15
      float dist = c2v - 2.0f * acc[v];
      if (dist < bv[v]) { bv[v] = dist; bi[v] = n0 + l15; }
    }
  }

  // cross-lane argmin within each 16-lane half (ties -> lowest index)
#pragma unroll
  for (int off = 1; off < 16; off <<= 1) {
#pragma unroll
    for (int v = 0; v < 8; ++v) {
      float ov = __shfl_xor(bv[v], off, 32);
      int   oi = __shfl_xor(bi[v], off, 32);
      if (ov < bv[v] || (ov == bv[v] && oi < bi[v])) { bv[v] = ov; bi[v] = oi; }
    }
  }

  // write indices: token = m0 + 8*half + v  -> [B, Q, T]
#pragma unroll
  for (int v = 0; v < 8; ++v) {
    if (l15 == v) {
      int tok = m0 + 8 * half + v;
      int b = tok >> 11, t = tok & 2047;
      idx_out[((size_t)b * RQ + stage) * RT + t] = (float)bi[v];
    }
  }

  // update residual (fp32, exact) + exact loss accumulation
  float lsum = 0.f;
#pragma unroll
  for (int tk = 0; tk < 16; ++tk) {
    int cidx = __shfl(bi[tk & 7], (tk >> 3) * 16, 32);
    const float4* crow = (const float4*)(cb_f32 + ((size_t)stage * RK + cidx) * RD);
    float4* rrow = (float4*)(residual + (size_t)(m0 + tk) * RD);
    float4 q = crow[lane];
    float4 r = rrow[lane];
    float dx = q.x - r.x, dy = q.y - r.y, dz = q.z - r.z, dw = q.w - r.w;
    lsum += dx * dx + dy * dy + dz * dz + dw * dw;
    rrow[lane] = make_float4(-dx, -dy, -dz, -dw);   // r - q
  }
#pragma unroll
  for (int off = 1; off < 32; off <<= 1) lsum += __shfl_xor(lsum, off, 32);
  if (lane == 0) wsum[wave] = lsum;
  __syncthreads();
  if (tid == 0) {   // deterministic ordered block sum
    float s = 0.f;
#pragma unroll
    for (int w = 0; w < 8; ++w) s += wsum[w];
    partials[stage * 256 + blockIdx.x] = s;
  }
}

// ---------------------------------------------------------------------------
// q_out = x - residual_final (same transpose mapping); one thread sums loss.
__global__ __launch_bounds__(256)
void rvq_finalize(const float* __restrict__ x, const float* __restrict__ res,
                  const float* __restrict__ part, float* __restrict__ out) {
  unsigned gid = blockIdx.x * 256u + threadIdx.x;
  unsigned b = gid >> 18, rem = gid & 262143u, d = rem >> 11, t = rem & 2047u;
  unsigned n = b * RT + t;
  out[gid] = x[gid] - res[(size_t)n * RD + d];
  if (gid == 0) {
    float s = 0.f;
    for (int i = 0; i < RQ * 256; ++i) s += part[i];   // fixed order -> deterministic
    out[(size_t)RB * RD * RT] =
        s * ((1.0f + RBETA) / ((float)RQ * (float)RN * (float)RD));
  }
}

// ---------------------------------------------------------------------------
extern "C" void kernel_launch(void* const* d_in, const int* in_sizes, int n_in,
                              void* d_out, int out_size, void* d_ws, size_t ws_size,
                              hipStream_t stream) {
  const float* x  = (const float*)d_in[0];   // [B, D, T] fp32
  const float* cb = (const float*)d_in[1];   // [Q, K, D] fp32
  float* out = (float*)d_out;                // q_out | loss | indices(as float)

  char*   ws       = (char*)d_ws;
  float*  residual = (float*)(ws + WS_RES);
  __bf16* cbhi     = (__bf16*)(ws + WS_CBHI);
  __bf16* cblo     = (__bf16*)(ws + WS_CBLO);
  float*  c2       = (float*)(ws + WS_C2);
  float*  partials = (float*)(ws + WS_PART);

  float* idx_out = out + (size_t)RB * RD * RT + 1;

  rvq_prep_res<<<(RN * RD) / 256, 256, 0, stream>>>(x, residual);
  rvq_prep_cb<<<(RQ * RK * 32) / 256, 256, 0, stream>>>(cb, cbhi, cblo, c2);

  for (int s = 0; s < RQ; ++s) {
    rvq_stage<<<RN / 128, 256, 0, stream>>>(residual, cbhi, cblo, c2, cb,
                                            partials, idx_out, s);
  }
  rvq_finalize<<<(RN * RD) / 256, 256, 0, stream>>>(x, residual, partials, out);
}
